// unit_gcn_12489764897489
// MI455X (gfx1250) — matrix operand order repaired and checked by reading, not compiled, not run on hardware
//
#include <hip/hip_runtime.h>
#include <hip/hip_bf16.h>

typedef __attribute__((ext_vector_type(16))) __bf16 v16bf;
typedef __attribute__((ext_vector_type(8)))  float  v8f;

#define N_ 128
#define C_ 64
#define T_ 128
#define V_ 25
#define S_ 3
#define R_ 8
#define O_ 64
#define TV_ (T_*V_)            // 3200
#define BN_EPS 1e-5f

// ---------------- Kernel 1: xm[n,c,v] = mean_t x[n,c,t,v] ----------------
__global__ void k_xm(const float* __restrict__ x, float* __restrict__ xm) {
    int i = blockIdx.x * 256 + threadIdx.x;
    if (i >= N_*C_*V_) return;
    int v  = i % V_;
    int nc = i / V_;
    const float* p = x + (size_t)nc * TV_ + v;
    float s = 0.f;
    for (int t = 0; t < T_; ++t) s += p[t * V_];
    xm[i] = s * (1.0f / T_);
}

// ---------------- Kernel 2: x1/x2[n,s,r,v] = xm · w + b ----------------
__global__ void k_x12(const float* __restrict__ xm,
                      const float* __restrict__ w1, const float* __restrict__ b1,
                      const float* __restrict__ w2, const float* __restrict__ b2,
                      float* __restrict__ x1, float* __restrict__ x2) {
    int i = blockIdx.x * 256 + threadIdx.x;
    if (i >= N_*S_*R_*V_) return;
    int v  = i % V_;
    int sr = (i / V_) % (S_ * R_);
    int n  = i / (S_ * R_ * V_);
    const float* xmp = xm + (size_t)n * C_ * V_ + v;
    const float* w1p = w1 + sr * C_;
    const float* w2p = w2 + sr * C_;
    float a = 0.f, b = 0.f;
    for (int c = 0; c < C_; ++c) {
        float xv = xmp[c * V_];
        a += xv * w1p[c];
        b += xv * w2p[c];
    }
    x1[i] = a + b1[sr];
    x2[i] = b + b2[sr];
}

// ---------------- Kernel 2b: prepack w3 into WMMA A-fragment layout ----------------
__global__ void k_w3p(const float* __restrict__ w3, __bf16* __restrict__ w3f) {
    int i = blockIdx.x * 256 + threadIdx.x;   // (s,og,ch,lane): 3*4*2*32 = 768
    if (i >= 768) return;
    int lane = i & 31;
    int ch   = (i >> 5) & 1;
    int og   = (i >> 6) & 3;
    int s    = i >> 8;
    int lmod = lane & 15, lhi = lane >> 4;
    int orow = og * 16 + lmod;
    #pragma unroll
    for (int h = 0; h < 16; ++h) {
        int c = ch*32 + h + 16*lhi;   // contiguous k layout: k = 16*lhi + h
        w3f[(size_t)i * 16 + h] = (__bf16)w3[(size_t)(s*O_ + orow) * C_ + c];
    }
}

// ---------------- Kernel 3: Mtf — M in WMMA B-fragment-ready bf16 layout ----------------
// logical M'[v][u] = alpha*(sum_r w4[s,o,r]*tanh(x1[n,s,r,u]-x2[n,s,r,v]) + b4[s,o]) + A[s,u,v]
// frag slot: [(n,s,o)]*1024 + (u>>4)*512 + ((v>>4)*16 + (u&15))*16 + (v&15)
__global__ void k_M(const float* __restrict__ x1, const float* __restrict__ x2,
                    const float* __restrict__ w4, const float* __restrict__ b4,
                    const float* __restrict__ A,  const float* __restrict__ alpha,
                    __bf16* __restrict__ Mtf) {
    int n = blockIdx.x, s = blockIdx.y;
    __shared__ float x1s[R_*V_], x2s[R_*V_], w4s[O_*R_], b4s[O_], As[V_*V_];
    int tid = threadIdx.x;
    for (int i = tid; i < R_*V_; i += 256) {
        x1s[i] = x1[(size_t)(n*S_+s)*R_*V_ + i];
        x2s[i] = x2[(size_t)(n*S_+s)*R_*V_ + i];
    }
    for (int i = tid; i < O_*R_;  i += 256) w4s[i] = w4[(size_t)s*O_*R_ + i];
    for (int i = tid; i < O_;     i += 256) b4s[i] = b4[s*O_ + i];
    for (int i = tid; i < V_*V_;  i += 256) As[i]  = A[(size_t)s*V_*V_ + i];
    __syncthreads();
    float al = alpha[0];
    __bf16* base = Mtf + (size_t)(n*S_+s) * O_ * 1024;
    for (int it = 0; it < 4; ++it) {
        int idx = it * 256 + tid;           // 0..1023 over (u,v) padded to 32x32
        int u = idx >> 5, v = idx & 31;
        size_t fofs = (size_t)(u >> 4)*512 + (size_t)((v >> 4)*16 + (u & 15))*16 + (v & 15);
        if (u < V_ && v < V_) {
            float att[R_];
            #pragma unroll
            for (int r = 0; r < R_; ++r) att[r] = tanhf(x1s[r*V_+u] - x2s[r*V_+v]);
            float Auv = As[u*V_+v];
            for (int o = 0; o < O_; ++o) {
                float m = 0.f;
                #pragma unroll
                for (int r = 0; r < R_; ++r) m += w4s[o*R_+r] * att[r];
                base[(size_t)o*1024 + fofs] = (__bf16)(al * (m + b4s[o]) + Auv);
            }
        } else {
            for (int o = 0; o < O_; ++o) base[(size_t)o*1024 + fofs] = (__bf16)0.f;
        }
    }
}

// ---------------- Kernel 3b: yb[n,o,u] = sum_s b3[s,o] * sum_v M'[n,s,o][v][u] ----------------
__global__ void k_yb(const __bf16* __restrict__ Mtf, const float* __restrict__ b3,
                     float* __restrict__ yb) {
    int i = blockIdx.x * 256 + threadIdx.x;
    if (i >= N_*O_*V_) return;
    int u = i % V_;
    int o = (i / V_) % O_;
    int n = i / (O_ * V_);
    int ut = u >> 4, lmod = u & 15;
    float acc = 0.f;
    for (int s = 0; s < S_; ++s) {
        const __bf16* m = Mtf + (size_t)((n*S_+s)*O_ + o) * 1024 + (size_t)ut*512;
        float rs = 0.f;
        for (int v = 0; v < V_; ++v)
            rs += (float)m[(size_t)((v >> 4)*16 + lmod)*16 + (v & 15)];
        acc += b3[s*O_ + o] * rs;
    }
    yb[i] = acc;
}

// ---------------- Kernel 4: fused x3-GEMM + graph aggregation (WMMA) ----------------
// block = 256 threads (8 wave32), grid = (N, T/16); all 4 o-groups handled per block
__global__ __launch_bounds__(256)
void k_main(const float* __restrict__ x, const __bf16* __restrict__ w3f,
            const __bf16* __restrict__ Mtf, const float* __restrict__ yb,
            float* __restrict__ y) {
    int n  = blockIdx.x;
    int t0 = blockIdx.y * 16;

    __shared__ __align__(32) __bf16 xs[400 * C_];       // TRANSPOSED: [j][c], j = t*25+v
    __shared__ __align__(32) __bf16 x3s[16 * 16 * V_];  // [o_local][t_local][v]

    int tid  = threadIdx.x;
    int lane = tid & 31;
    int wave = __builtin_amdgcn_readfirstlane(tid >> 5);   // scalar -> uniform loops
    int lmod = lane & 15, lhi = lane >> 4;

    // stage x[n, :, t0:t0+16, :] -> bf16 LDS, transposed so B1 fragments are contiguous in c
    const float* xn = x + (size_t)n * C_ * TV_ + (size_t)t0 * V_;
    for (int i = tid; i < C_ * 400; i += 256) {
        int c = i / 400, j = i % 400;                   // global reads coalesced over j
        xs[j * C_ + c] = (__bf16)xn[(size_t)c * TV_ + j];
    }
    __syncthreads();

    v8f vzero = {0.f,0.f,0.f,0.f,0.f,0.f,0.f,0.f};
    v8f acc[4][2][2];                                   // [og][p][ut]
    #pragma unroll
    for (int og = 0; og < 4; ++og)
        #pragma unroll
        for (int p = 0; p < 2; ++p) { acc[og][p][0] = vzero; acc[og][p][1] = vzero; }

    for (int s = 0; s < S_; ++s) {
        #pragma unroll
        for (int og = 0; og < 4; ++og) {
            // warm the near cache with this (s,og)'s Mt fragment slab (32KB = 256 lines)
            const __bf16* mslab = Mtf + ((size_t)((n*S_+s)*O_) + og*16) * 1024;
            __builtin_prefetch((const char*)mslab + tid * 128, 0, 1);

            // A fragments for GEMM-1 (prepacked): one 32B vector load per chunk
            v16bf a1[2];
            #pragma unroll
            for (int ch = 0; ch < 2; ++ch)
                a1[ch] = *(const v16bf*)&w3f[(size_t)(((s*4 + og)*2 + ch)*32 + lane) * 16];

            // GEMM-1: x3[o,(t,v)] = sum_c w3[s,o,c]*x[n,c,t,v]; 25 col-tiles over 8 waves
            for (int jt = wave; jt < 25; jt += 8) {
                v8f c1 = vzero;
                int jcol = jt*16 + lmod;
                #pragma unroll
                for (int ch = 0; ch < 2; ++ch) {
                    v16bf bf = *(const v16bf*)&xs[jcol * C_ + ch*32 + 16*lhi];
                    c1 = __builtin_amdgcn_wmma_f32_16x16x32_bf16(
                            false, a1[ch], false, bf, (short)0, c1, false, false);
                }
                int tt = jcol / 25, vv = jcol % 25;
                #pragma unroll
                for (int r = 0; r < 8; ++r) {
                    int ol = r + 8*lhi;                 // C/D layout: m = r + 8*(lane>>4)
                    x3s[(ol*16 + tt)*V_ + vv] = (__bf16)c1[r];
                }
            }
            __syncthreads();
            // GEMM-2: y[t][u] += sum_v x3[o][t][v] * M'[n,s,o][v][u]; 2 o's per wave
            #pragma unroll
            for (int p = 0; p < 2; ++p) {
                int ol = wave*2 + p;
                int o  = og*16 + ol;
                v16bf a2;
                #pragma unroll
                for (int h = 0; h < 16; ++h) {
                    int v = h + 16*lhi;     // k (=v) index 0..31
                    a2[h] = (v < V_) ? x3s[(ol*16 + lmod)*V_ + v] : (__bf16)0.f;
                }
                const __bf16* mb = Mtf + (size_t)((n*S_+s)*O_ + o) * 1024;
                #pragma unroll
                for (int ut = 0; ut < 2; ++ut) {
                    v16bf b2 = *(const v16bf*)&mb[(size_t)ut*512 + (size_t)lane*16];
                    acc[og][p][ut] = __builtin_amdgcn_wmma_f32_16x16x32_bf16(
                            false, a2, false, b2, (short)0, acc[og][p][ut], false, false);
                }
            }
            __syncthreads();   // protect x3s before next (s,og) rewrites it
        }
    }

    // --- write y[n,o,t,u] (+ folded b3 term yb[n,o,u])
    #pragma unroll
    for (int og = 0; og < 4; ++og) {
        #pragma unroll
        for (int p = 0; p < 2; ++p) {
            int o = og*16 + wave*2 + p;
            const float* ybp = yb + ((size_t)n*O_ + o) * V_;
            #pragma unroll
            for (int ut = 0; ut < 2; ++ut) {
                int u = ut*16 + lmod;
                if (u < V_) {
                    #pragma unroll
                    for (int r = 0; r < 8; ++r) {
                        int tt = r + 8*lhi;
                        y[(((size_t)n*O_ + o)*T_ + t0 + tt)*V_ + u] = acc[og][p][ut][r] + ybp[u];
                    }
                }
            }
        }
    }
}

// ---------------- Kernel 5: per-channel batch stats ----------------
__global__ void k_stats(const float* __restrict__ y, float* __restrict__ stats) {
    int o = blockIdx.x;
    int tid = threadIdx.x;
    float s1 = 0.f, s2 = 0.f;
    for (int i = tid; i < N_*T_*V_; i += 256) {
        int n = i / TV_, j = i % TV_;
        float v = y[((size_t)n*O_ + o) * TV_ + j];
        s1 += v; s2 += v*v;
    }
    __shared__ float r1[256], r2[256];
    r1[tid] = s1; r2[tid] = s2;
    __syncthreads();
    for (int st = 128; st > 0; st >>= 1) {
        if (tid < st) { r1[tid] += r1[tid+st]; r2[tid] += r2[tid+st]; }
        __syncthreads();
    }
    if (tid == 0) {
        float cnt  = (float)(N_*T_*V_);
        float mean = r1[0] / cnt;
        float var  = r2[0] / cnt - mean*mean;   // biased, like jnp.var
        stats[o]      = mean;
        stats[O_ + o] = rsqrtf(var + BN_EPS);
    }
}

// ---------------- Kernel 6: BN + residual + ReLU (in place on d_out) ----------------
__global__ void k_apply(float* y, const float* __restrict__ x,
                        const float* __restrict__ stats,
                        const float* __restrict__ bnw, const float* __restrict__ bnb) {
    int i = blockIdx.x * 256 + threadIdx.x;
    if (i >= N_*O_*T_*V_) return;
    int o = (i / TV_) % O_;
    float val = (y[i] - stats[o]) * stats[O_+o] * bnw[o] + bnb[o] + x[i];
    y[i] = fmaxf(val, 0.f);
}

extern "C" void kernel_launch(void* const* d_in, const int* in_sizes, int n_in,
                              void* d_out, int out_size, void* d_ws, size_t ws_size,
                              hipStream_t stream) {
    (void)in_sizes; (void)n_in; (void)out_size; (void)ws_size;
    const float* x     = (const float*)d_in[0];
    const float* A     = (const float*)d_in[1];
    const float* alpha = (const float*)d_in[2];
    const float* w1    = (const float*)d_in[3];
    const float* b1    = (const float*)d_in[4];
    const float* w2    = (const float*)d_in[5];
    const float* b2    = (const float*)d_in[6];
    const float* w3    = (const float*)d_in[7];
    const float* b3    = (const float*)d_in[8];
    const float* w4    = (const float*)d_in[9];
    const float* b4    = (const float*)d_in[10];
    const float* bnw   = (const float*)d_in[11];
    const float* bnb   = (const float*)d_in[12];
    float* y = (float*)d_out;

    char* ws = (char*)d_ws;                    // ~52.6 MB used
    float*  xm  = (float*)(ws);                // 819,200 B
    float*  x1  = (float*)(ws + 819200);       // 307,200 B
    float*  x2  = (float*)(ws + 1126400);      // 307,200 B
    float*  yb  = (float*)(ws + 1433600);      // 819,200 B
    float*  st  = (float*)(ws + 2252800);      // 512 B
    __bf16* w3f = (__bf16*)(ws + 2253312);     // 24,576 B  (32B-aligned)
    __bf16* Mtf = (__bf16*)(ws + 2277888);     // 50,331,648 B (32B-aligned)

    k_xm   <<<800,  256, 0, stream>>>(x, xm);
    k_x12  <<<300,  256, 0, stream>>>(xm, w1, b1, w2, b2, x1, x2);
    k_w3p  <<<3,    256, 0, stream>>>(w3, w3f);
    k_M    <<<dim3(N_, S_), 256, 0, stream>>>(x1, x2, w4, b4, A, alpha, Mtf);
    k_yb   <<<800,  256, 0, stream>>>(Mtf, b3, yb);
    k_main <<<dim3(N_, T_/16), 256, 0, stream>>>(x, w3f, Mtf, yb, y);
    k_stats<<<O_,   256, 0, stream>>>(y, st);
    k_apply<<<40960,256, 0, stream>>>(y, x, st, bnw, bnb);
}
